// SetMatchingLoss_80968723464872
// MI455X (gfx1250) — matrix operand order repaired and checked by reading, not compiled, not run on hardware
//
#include <hip/hip_runtime.h>

typedef __attribute__((ext_vector_type(16))) _Float16 v16h;
typedef __attribute__((ext_vector_type(8)))  _Float16 v8h;
typedef __attribute__((ext_vector_type(8)))  float    v8f;

#define BQ   64
#define NQ   128
#define ND   512
#define FEPS 1e-8f
#define FINF 1e9f

// ---------------------------------------------------------------------------
// Kernel 1: row-normalize (x + eps) / max(||x + eps||, eps) and cast to f16.
// One block (128 threads) per row of 512.
// ---------------------------------------------------------------------------
__global__ void nrm_f16_kernel(const float* __restrict__ x,
                               _Float16* __restrict__ y) {
    const int row = blockIdx.x;
    const int t   = threadIdx.x;
    const float* xr = x + (size_t)row * ND;
    float vals[4];
    float s = 0.f;
#pragma unroll
    for (int k = 0; k < 4; ++k) {
        float v = xr[t + 128 * k] + FEPS;
        vals[k] = v;
        s += v * v;
    }
    __shared__ float red[128];
    red[t] = s;
    __syncthreads();
    for (int st = 64; st > 0; st >>= 1) {
        if (t < st) red[t] += red[t + st];
        __syncthreads();
    }
    const float inv = 1.0f / fmaxf(sqrtf(red[0]), FEPS);
#pragma unroll
    for (int k = 0; k < 4; ++k)
        y[(size_t)row * ND + t + 128 * k] = (_Float16)(vals[k] * inv);
}

// ---------------------------------------------------------------------------
// Kernel 2: batched GEMM cost = -(r @ e^T) via v_wmma_f32_16x16x32_f16.
// Grid: (64 tiles, 64 batches), one wave (32 threads) per 16x16 tile.
// ---------------------------------------------------------------------------
__global__ void wmma_cost_kernel(const _Float16* __restrict__ r16,
                                 const _Float16* __restrict__ e16,
                                 float* __restrict__ cost) {
    const int lane = threadIdx.x;        // 0..31, EXEC all-ones for WMMA
    const int half = lane >> 4;          // 0/1
    const int lm   = lane & 15;
    const int b    = blockIdx.y;
    const int mt   = blockIdx.x >> 3;    // tile row 0..7
    const int nt   = blockIdx.x & 7;     // tile col 0..7

    const _Float16* ra = r16 + ((size_t)(b * NQ + mt * 16 + lm)) * ND;
    const _Float16* rb = e16 + ((size_t)(b * NQ + nt * 16 + lm)) * ND;

    v8f acc = {};
#pragma unroll 4
    for (int kb = 0; kb < ND / 32; ++kb) {
        // A fragment: two 8-half runs at K = kb*32 + half*16 and +8
        const int kA = kb * 32 + half * 16;
        union { v16h v; v8h h[2]; } av;
        av.h[0] = *(const v8h*)(ra + kA);
        av.h[1] = *(const v8h*)(ra + kA + 8);
        // B fragment: 16 contiguous halves at K = kb*32 + half*16 (32B aligned)
        const v16h bv = *(const v16h*)(rb + kb * 32 + half * 16);
        acc = __builtin_amdgcn_wmma_f32_16x16x32_f16(
            /*neg_a=*/false, av.v, /*neg_b=*/false, bv,
            /*c_mod=*/(short)0, acc, /*reuse_a=*/false, /*reuse_b=*/false);
    }

    float* cb = cost + (size_t)b * NQ * NQ;
#pragma unroll
    for (int v = 0; v < 8; ++v) {
        const int m = mt * 16 + half * 8 + v;
        const int n = nt * 16 + lm;
        cb[m * NQ + n] = -acc[v];   // cost = negative cosine similarity
    }
}

// ---------------------------------------------------------------------------
// Kernel 3: exact Jonker-Volgenant LSA, one block (128 threads) per batch.
// Mirrors the reference (1-indexed with virtual row/col 0, first-index
// tie-break argmin). State in LDS; cost rows streamed from L2.
// ---------------------------------------------------------------------------
__global__ void lsa_kernel(const float* __restrict__ cost,
                           int* __restrict__ colInt,
                           float* __restrict__ colOut) {
    const int b = blockIdx.x;
    const int t = threadIdx.x;                  // 0..127
    const float* C = cost + (size_t)b * NQ * NQ;

    __shared__ float u[NQ + 1], v[NQ + 1], minv[NQ + 1];
    __shared__ int   p[NQ + 1], way[NQ + 1];
    __shared__ unsigned char used[NQ + 1];
    __shared__ float redV[128];
    __shared__ int   redI[128];
    __shared__ int   sj0;

    for (int j = t; j <= NQ; j += 128) { u[j] = 0.f; v[j] = 0.f; p[j] = 0; }
    __syncthreads();

    for (int i = 1; i <= NQ; ++i) {
        if (t == 0) { p[0] = i; sj0 = 0; }
        for (int j = t; j <= NQ; j += 128) {
            minv[j] = FINF; way[j] = 0; used[j] = 0;
        }
        __syncthreads();

        while (true) {
            const int j0 = sj0;
            const int i0 = p[j0];
            if (i0 == 0) break;              // uniform across block
            if (t == 0) used[j0] = 1;
            __syncthreads();

            const float ui0 = u[i0];
            for (int j = t; j <= NQ; j += 128) {
                if (!used[j]) {
                    const float cj  = (j == 0) ? 0.f : C[(i0 - 1) * NQ + (j - 1)];
                    const float cur = cj - ui0 - v[j];
                    if (cur < minv[j]) { minv[j] = cur; way[j] = j0; }
                }
            }
            __syncthreads();

            // argmin over masked minv, first-index tie-break (== jnp.argmin)
            float bv = FINF; int bi = NQ + 1;
            for (int j = t; j <= NQ; j += 128) {
                const float m = used[j] ? FINF : minv[j];
                if (m < bv || (m == bv && j < bi)) { bv = m; bi = j; }
            }
            redV[t] = bv; redI[t] = bi;
            __syncthreads();
            for (int st = 64; st > 0; st >>= 1) {
                if (t < st) {
                    const float ov = redV[t + st]; const int oi = redI[t + st];
                    if (ov < redV[t] || (ov == redV[t] && oi < redI[t])) {
                        redV[t] = ov; redI[t] = oi;
                    }
                }
                __syncthreads();
            }
            const int   j1    = redI[0];
            const float delta = redV[0];

            for (int j = t; j <= NQ; j += 128) {
                if (used[j]) {
                    u[p[j]] += delta;        // p[] distinct over used columns
                    v[j]    -= delta;
                } else {
                    minv[j] -= delta;
                }
            }
            if (t == 0) sj0 = j1;
            __syncthreads();
        }

        if (t == 0) {                        // augment alternating path
            int j0 = sj0;
            while (j0 != 0) { const int j1 = way[j0]; p[j0] = p[j1]; j0 = j1; }
        }
        __syncthreads();
    }

    // col_ind[p[j]-1] = j-1
    {
        const int j = t + 1;
        const int row = p[j] - 1;
        colInt[b * NQ + row] = j - 1;
        colOut[b * NQ + row] = (float)(j - 1);
    }
}

// ---------------------------------------------------------------------------
// Kernel 4: exact f32 cosine similarity for matched pairs.
// One block (128 threads) per (b, i).
// ---------------------------------------------------------------------------
__global__ void sim_kernel(const float* __restrict__ r,
                           const float* __restrict__ e,
                           const int* __restrict__ colInt,
                           float* __restrict__ sim) {
    const int idx = blockIdx.x;              // b*NQ + i
    const int b   = idx >> 7;
    const int c   = colInt[idx];
    const int t   = threadIdx.x;
    const float* rr = r + (size_t)idx * ND;
    const float* ee = e + (size_t)(b * NQ + c) * ND;

    float sr = 0.f, se = 0.f, sd = 0.f;
#pragma unroll
    for (int k = 0; k < 4; ++k) {
        const float a  = rr[t + 128 * k] + FEPS;
        const float bb = ee[t + 128 * k] + FEPS;
        sr += a * a; se += bb * bb; sd += a * bb;
    }
    __shared__ float R0[128], R1[128], R2[128];
    R0[t] = sr; R1[t] = se; R2[t] = sd;
    __syncthreads();
    for (int st = 64; st > 0; st >>= 1) {
        if (t < st) { R0[t] += R0[t + st]; R1[t] += R1[t + st]; R2[t] += R2[t + st]; }
        __syncthreads();
    }
    if (t == 0) {
        const float nr = fmaxf(sqrtf(R0[0]), FEPS);
        const float ne = fmaxf(sqrtf(R1[0]), FEPS);
        sim[idx] = R2[0] / (nr * ne);
    }
}

// ---------------------------------------------------------------------------
// Kernel 5: deterministic final reduction. loss = 1 - mean(sim).
// ---------------------------------------------------------------------------
__global__ void loss_kernel(const float* __restrict__ sim, float* __restrict__ out) {
    __shared__ float red[256];
    const int t = threadIdx.x;
    float s = 0.f;
    for (int k = t; k < BQ * NQ; k += 256) s += sim[k];
    red[t] = s;
    __syncthreads();
    for (int st = 128; st > 0; st >>= 1) {
        if (t < st) red[t] += red[t + st];
        __syncthreads();
    }
    if (t == 0) out[0] = 1.0f - red[0] / (float)(BQ * NQ);
}

// ---------------------------------------------------------------------------
extern "C" void kernel_launch(void* const* d_in, const int* in_sizes, int n_in,
                              void* d_out, int out_size, void* d_ws, size_t ws_size,
                              hipStream_t stream) {
    const float* rubin  = (const float*)d_in[0];
    const float* euclid = (const float*)d_in[1];
    float* out = (float*)d_out;              // [0]=loss, [1..8192]=col (as float)

    char* ws = (char*)d_ws;
    size_t off = 0;
    _Float16* r16  = (_Float16*)(ws + off); off += (size_t)BQ * NQ * ND * sizeof(_Float16);
    _Float16* e16  = (_Float16*)(ws + off); off += (size_t)BQ * NQ * ND * sizeof(_Float16);
    float*    cost = (float*)   (ws + off); off += (size_t)BQ * NQ * NQ * sizeof(float);
    int*      col  = (int*)     (ws + off); off += (size_t)BQ * NQ * sizeof(int);
    float*    sim  = (float*)   (ws + off); off += (size_t)BQ * NQ * sizeof(float);
    (void)ws_size; (void)in_sizes; (void)n_in; (void)out_size;

    nrm_f16_kernel<<<BQ * NQ, 128, 0, stream>>>(rubin,  r16);
    nrm_f16_kernel<<<BQ * NQ, 128, 0, stream>>>(euclid, e16);
    wmma_cost_kernel<<<dim3(64, BQ), 32, 0, stream>>>(r16, e16, cost);
    lsa_kernel<<<BQ, 128, 0, stream>>>(cost, col, out + 1);
    sim_kernel<<<BQ * NQ, 128, 0, stream>>>(rubin, euclid, col, sim);
    loss_kernel<<<1, 256, 0, stream>>>(sim, out);
}
